// SparseMoE_44890998177817
// MI455X (gfx1250) — matrix operand order repaired and checked by reading, not compiled, not run on hardware
//
#include <hip/hip_runtime.h>

#define H   1024
#define FF  4096
#define NE  8
#define T   8192
#define FC  128                 // FF chunk per fused iteration
#define MT  32                  // token tile (packed path): 2x16 WMMA M-tiles
#define MAXT32 (T / MT)         // 256
#define MT16 16                 // token tile (fallback path)
#define MAXT16 (T / MT16)       // 512
#define XS  (H + 8)             // padded LDS row stride for X planes
#define HS  (FC + 8)            // padded LDS row stride for hidden planes

typedef __attribute__((ext_vector_type(16))) __bf16          v16bf;
typedef __attribute__((ext_vector_type(16))) unsigned short  v16u;
typedef __attribute__((ext_vector_type(8)))  unsigned short  v8us;
typedef __attribute__((ext_vector_type(8)))  float           v8f;

// round-to-nearest-even bf16, returned as f32 bit pattern (low 16 bits zero)
__device__ __forceinline__ unsigned bf16_rne(float f) {
  unsigned u = __float_as_uint(f);
  return (u + 0x7fffu + ((u >> 16) & 1u)) & 0xffff0000u;
}
// pack {hi,lo} split of f into one u32 (fallback path)
__device__ __forceinline__ unsigned pack_split(float f) {
  unsigned hib = bf16_rne(f);
  unsigned lob = bf16_rne(f - __uint_as_float(hib));
  return hib | (lob >> 16);
}
// A-fragment: two 8-element runs 16 apart (K = 8g..8g+8 and 16+8g..+8)
__device__ __forceinline__ v16u load_frag16(const unsigned short* p) {
  v8us a = *(const v8us*)(p);
  v8us b = *(const v8us*)(p + 16);
  return __builtin_shufflevector(a, b, 0, 1, 2, 3, 4, 5, 6, 7,
                                 8, 9, 10, 11, 12, 13, 14, 15);
}
#define WMMA_BF16(AH, BH, C)                                              \
  __builtin_amdgcn_wmma_f32_16x16x32_bf16(false, __builtin_bit_cast(v16bf, AH), \
                                          false, __builtin_bit_cast(v16bf, BH), \
                                          (short)0, (C), false, false)

// ---------------------------------------------------------------- init ----
__global__ void moe_init(float* __restrict__ out, int n, int* __restrict__ counts) {
  int i = blockIdx.x * blockDim.x + threadIdx.x;
  if (i < n)  out[i] = 0.f;
  if (i < NE) counts[i] = 0;
}

// ---------------------------------------------------------------- gate ----
__global__ void moe_gate(const float* __restrict__ x, const float* __restrict__ Wg,
                         const float* __restrict__ bg, float* __restrict__ logits,
                         int* __restrict__ counts, int* __restrict__ tok_idx,
                         float* __restrict__ tok_w) {
  const int wave = threadIdx.x >> 5;
  const int lane = threadIdx.x & 31;
  const int t = blockIdx.x * 8 + wave;
  if (t >= T) return;

  float acc[NE];
#pragma unroll
  for (int e = 0; e < NE; ++e) acc[e] = 0.f;

  const float* xr = x + (size_t)t * H;
  for (int h = lane; h < H; h += 32) {
    float xv = xr[h];
#pragma unroll
    for (int e = 0; e < NE; ++e) acc[e] = fmaf(xv, Wg[h * NE + e], acc[e]);
  }
#pragma unroll
  for (int e = 0; e < NE; ++e) {
#pragma unroll
    for (int off = 16; off > 0; off >>= 1) acc[e] += __shfl_xor(acc[e], off, 32);
  }

  if (lane == 0) {
    float lg[NE], p[NE];
    float mx = -1e30f;
#pragma unroll
    for (int e = 0; e < NE; ++e) {
      lg[e] = acc[e] + bg[e];
      logits[t * NE + e] = lg[e];
      mx = fmaxf(mx, lg[e]);
    }
    float s = 0.f;
#pragma unroll
    for (int e = 0; e < NE; ++e) { p[e] = expf(lg[e] - mx); s += p[e]; }
    float inv = 1.f / s;
#pragma unroll
    for (int e = 0; e < NE; ++e) p[e] *= inv;

    int i0 = 0;
#pragma unroll
    for (int e = 1; e < NE; ++e) if (p[e] > p[i0]) i0 = e;
    int i1 = (i0 == 0) ? 1 : 0;
#pragma unroll
    for (int e = 0; e < NE; ++e) if (e != i0 && p[e] > p[i1]) i1 = e;

    float m2 = fmaxf(p[i0], p[i1]);
    float e0 = expf(p[i0] - m2), e1 = expf(p[i1] - m2);
    float inv2 = 1.f / (e0 + e1);

    int pos0 = atomicAdd(&counts[i0], 1);
    tok_idx[i0 * T + pos0] = t;  tok_w[i0 * T + pos0] = e0 * inv2;
    int pos1 = atomicAdd(&counts[i1], 1);
    tok_idx[i1 * T + pos1] = t;  tok_w[i1 * T + pos1] = e1 * inv2;
  }
}

// ---------------------------------------------------------------- pack ----
// Convert fp32 weights into split-bf16 hi/lo planes, stored per 32x16 tile in
// B-fragment order: [tile][lane][16 K in table order] with lane = 16*(K>>3 & 1)+col.
__global__ void moe_pack(const float* __restrict__ W1, const float* __restrict__ W2,
                         unsigned short* __restrict__ W1hi, unsigned short* __restrict__ W1lo,
                         unsigned short* __restrict__ W2hi, unsigned short* __restrict__ W2lo) {
  const size_t n = (size_t)NE * H * FF;
  for (size_t idx = (size_t)blockIdx.x * blockDim.x + threadIdx.x; idx < n;
       idx += (size_t)gridDim.x * blockDim.x) {
    {
      size_t r = idx;
      int nn = (int)(r % FF); r /= FF;
      int K  = (int)(r % H);  r /= H;
      int e  = (int)r;
      float f = W1[idx];
      unsigned hib = bf16_rne(f);
      unsigned lob = bf16_rne(f - __uint_as_float(hib));
      int kt = K >> 5, kr = K & 31, g = (kr >> 3) & 1, i = (kr & 7) + 8 * (kr >> 4);
      int nt = nn >> 4, col = nn & 15, lane = g * 16 + col;
      size_t d = ((((size_t)(e * (H / 32) + kt)) * (FF / 16) + nt) * 32 + lane) * 16 + i;
      W1hi[d] = (unsigned short)(hib >> 16);
      W1lo[d] = (unsigned short)(lob >> 16);
    }
    {
      size_t r = idx;
      int nn = (int)(r % H);  r /= H;
      int K  = (int)(r % FF); r /= FF;
      int e  = (int)r;
      float f = W2[idx];
      unsigned hib = bf16_rne(f);
      unsigned lob = bf16_rne(f - __uint_as_float(hib));
      int kt = K >> 5, kr = K & 31, g = (kr >> 3) & 1, i = (kr & 7) + 8 * (kr >> 4);
      int nt = nn >> 4, col = nn & 15, lane = g * 16 + col;
      size_t d = ((((size_t)(e * (FF / 32) + kt)) * (H / 16) + nt) * 32 + lane) * 16 + i;
      W2hi[d] = (unsigned short)(hib >> 16);
      W2lo[d] = (unsigned short)(lob >> 16);
    }
  }
}

// ---------------------------------------------------- ffn (packed path) ----
// One workgroup = one expert, 32 routed tokens (2x16 M-tiles); 8 waves.
__global__ void __launch_bounds__(256, 1)
moe_ffn_packed(const float* __restrict__ x,
               const unsigned short* __restrict__ W1hi, const unsigned short* __restrict__ W1lo,
               const unsigned short* __restrict__ W2hi, const unsigned short* __restrict__ W2lo,
               const float* __restrict__ b1, const float* __restrict__ b2,
               const int* __restrict__ counts, const int* __restrict__ tok_idx,
               const float* __restrict__ tok_w, float* __restrict__ out) {
  __shared__ unsigned short sXhi[MT * XS], sXlo[MT * XS];   // ~66 KB each
  __shared__ unsigned short sHhi[MT * HS], sHlo[MT * HS];   // ~8.7 KB each
  __shared__ int   sTok[MT];
  __shared__ float sW[MT];

  const int e    = blockIdx.x >> 8;           // / MAXT32
  const int mt   = blockIdx.x & (MAXT32 - 1);
  const int n_e  = counts[e];
  const int base = mt * MT;
  if (base >= n_e) return;

  const int tid  = threadIdx.x;
  const int wave = tid >> 5;
  const int lane = tid & 31;
  const int ln   = lane & 15;
  const int g    = lane >> 4;

  if (tid < MT) {
    int p = base + tid;
    if (p < n_e) { sTok[tid] = tok_idx[e * T + p]; sW[tid] = tok_w[e * T + p]; }
    else         { sTok[tid] = -1;                 sW[tid] = 0.f; }
  }
  __syncthreads();

  for (int i = tid; i < MT * H; i += 256) {
    int m = i >> 10;
    int k = i & (H - 1);
    int tkn = sTok[m];
    float f = (tkn >= 0) ? x[(size_t)tkn * H + k] : 0.f;
    unsigned hib = bf16_rne(f);
    unsigned lob = bf16_rne(f - __uint_as_float(hib));
    sXhi[m * XS + k] = (unsigned short)(hib >> 16);
    sXlo[m * XS + k] = (unsigned short)(lob >> 16);
  }
  __syncthreads();

  v8f acc0[8], acc1[8];
#pragma unroll
  for (int i = 0; i < 8; ++i) { acc0[i] = v8f{}; acc1[i] = v8f{}; }

  for (int ff0 = 0; ff0 < FF; ff0 += FC) {
    // ---- GEMM1: H1[:, ff0 + wave*16 + (0..15)] for both 16-row halves ----
    v8f c0 = v8f{}, c1 = v8f{};
    const int nt1 = (ff0 >> 4) + wave;
#pragma unroll 2
    for (int kt = 0; kt < H / 32; ++kt) {
      v16u ah0 = load_frag16(&sXhi[ln * XS + kt * 32 + 8 * g]);
      v16u al0 = load_frag16(&sXlo[ln * XS + kt * 32 + 8 * g]);
      v16u ah1 = load_frag16(&sXhi[(16 + ln) * XS + kt * 32 + 8 * g]);
      v16u al1 = load_frag16(&sXlo[(16 + ln) * XS + kt * 32 + 8 * g]);
      size_t off = ((((size_t)(e * (H / 32) + kt)) * (FF / 16) + nt1) * 32 + lane) * 16;
      v16u bh = *(const v16u*)(W1hi + off);
      v16u bl = *(const v16u*)(W1lo + off);
      c0 = WMMA_BF16(ah0, bh, c0);
      c0 = WMMA_BF16(ah0, bl, c0);
      c0 = WMMA_BF16(al0, bh, c0);
      c1 = WMMA_BF16(ah1, bh, c1);
      c1 = WMMA_BF16(ah1, bl, c1);
      c1 = WMMA_BF16(al1, bh, c1);
    }
    {
      const int col1 = ff0 + wave * 16 + ln;
      float bias = b1[e * FF + col1];
#pragma unroll
      for (int r = 0; r < 8; ++r) {              // C/D: M = r + 8g, N = ln
        int M = r + 8 * g;
        float v0 = fmaxf(c0[r] + bias, 0.f);
        float v1 = fmaxf(c1[r] + bias, 0.f);
        unsigned h0 = bf16_rne(v0), h1 = bf16_rne(v1);
        unsigned l0 = bf16_rne(v0 - __uint_as_float(h0));
        unsigned l1 = bf16_rne(v1 - __uint_as_float(h1));
        sHhi[M * HS + wave * 16 + ln]        = (unsigned short)(h0 >> 16);
        sHlo[M * HS + wave * 16 + ln]        = (unsigned short)(l0 >> 16);
        sHhi[(16 + M) * HS + wave * 16 + ln] = (unsigned short)(h1 >> 16);
        sHlo[(16 + M) * HS + wave * 16 + ln] = (unsigned short)(l1 >> 16);
      }
    }
    __syncthreads();

    // ---- GEMM2: out cols [wave*128, wave*128+128), K-chunk ff0..ff0+FC ----
#pragma unroll 1
    for (int kk = 0; kk < FC; kk += 32) {
      v16u ah0 = load_frag16(&sHhi[ln * HS + kk + 8 * g]);
      v16u al0 = load_frag16(&sHlo[ln * HS + kk + 8 * g]);
      v16u ah1 = load_frag16(&sHhi[(16 + ln) * HS + kk + 8 * g]);
      v16u al1 = load_frag16(&sHlo[(16 + ln) * HS + kk + 8 * g]);
      const int kt2 = (ff0 + kk) >> 5;
#pragma unroll
      for (int tt = 0; tt < 8; ++tt) {
        int nt2 = wave * 8 + tt;
        size_t off = ((((size_t)(e * (FF / 32) + kt2)) * (H / 16) + nt2) * 32 + lane) * 16;
        v16u bh = *(const v16u*)(W2hi + off);
        v16u bl = *(const v16u*)(W2lo + off);
        acc0[tt] = WMMA_BF16(ah0, bh, acc0[tt]);
        acc0[tt] = WMMA_BF16(ah0, bl, acc0[tt]);
        acc0[tt] = WMMA_BF16(al0, bh, acc0[tt]);
        acc1[tt] = WMMA_BF16(ah1, bh, acc1[tt]);
        acc1[tt] = WMMA_BF16(ah1, bl, acc1[tt]);
        acc1[tt] = WMMA_BF16(al1, bh, acc1[tt]);
      }
    }
    __syncthreads();   // sH rewritten next chunk
  }

#pragma unroll
  for (int tt = 0; tt < 8; ++tt) {
    int col = wave * 128 + tt * 16 + ln;
    float bias = b2[e * H + col];
#pragma unroll
    for (int r = 0; r < 8; ++r) {
      int M = r + 8 * g;
      int t0 = sTok[M], t1 = sTok[16 + M];
      if (t0 >= 0) atomicAdd(&out[(size_t)t0 * H + col], (acc0[tt][r] + bias) * sW[M]);
      if (t1 >= 0) atomicAdd(&out[(size_t)t1 * H + col], (acc1[tt][r] + bias) * sW[16 + M]);
    }
  }
}

// ----------------------------------------- ffn (fallback, convert-in-loop) --
__global__ void __launch_bounds__(256, 1)
moe_ffn_inline(const float* __restrict__ x,
               const float* __restrict__ W1, const float* __restrict__ b1,
               const float* __restrict__ W2, const float* __restrict__ b2,
               const int* __restrict__ counts, const int* __restrict__ tok_idx,
               const float* __restrict__ tok_w, float* __restrict__ out) {
  __shared__ unsigned sX[MT16 * H];
  __shared__ unsigned sH[MT16 * FC];
  __shared__ int      sTok[MT16];
  __shared__ float    sW[MT16];

  const int e    = blockIdx.x >> 9;
  const int mt   = blockIdx.x & (MAXT16 - 1);
  const int n_e  = counts[e];
  const int base = mt * MT16;
  if (base >= n_e) return;

  const int tid  = threadIdx.x;
  const int wave = tid >> 5;
  const int lane = tid & 31;
  const int ln   = lane & 15;
  const int g    = lane >> 4;

  if (tid < MT16) {
    int p = base + tid;
    if (p < n_e) { sTok[tid] = tok_idx[e * T + p]; sW[tid] = tok_w[e * T + p]; }
    else         { sTok[tid] = -1;                 sW[tid] = 0.f; }
  }
  __syncthreads();

  for (int i = tid; i < MT16 * H; i += 256) {
    int m = i >> 10;
    int k = i & (H - 1);
    int tkn = sTok[m];
    float f = (tkn >= 0) ? x[(size_t)tkn * H + k] : 0.f;
    sX[i] = pack_split(f);
  }
  __syncthreads();

  const float* W1p = W1 + (size_t)e * H * FF;
  const float* W2p = W2 + (size_t)e * FF * H;

  v8f acc[8];
#pragma unroll
  for (int i = 0; i < 8; ++i) acc[i] = v8f{};

  for (int ff0 = 0; ff0 < FF; ff0 += FC) {
    v8f c1 = v8f{};
    const int col1 = ff0 + wave * 16 + ln;
#pragma unroll 1
    for (int k0 = 0; k0 < H; k0 += 32) {
      v16u ah, al, bh, bl;
#pragma unroll
      for (int i = 0; i < 8; ++i) {
        int K = k0 + 16 * g + 2 * i;
        unsigned w0 = sX[ln * H + K];
        unsigned w1 = sX[ln * H + K + 1];
        ah[2 * i] = (unsigned short)(w0 >> 16); al[2 * i] = (unsigned short)(w0 & 0xffffu);
        ah[2 * i + 1] = (unsigned short)(w1 >> 16); al[2 * i + 1] = (unsigned short)(w1 & 0xffffu);
      }
#pragma unroll
      for (int j = 0; j < 8; ++j) {
        int K = k0 + 16 * g + 2 * j;
        float f0 = W1p[(size_t)K * FF + col1];
        float f1 = W1p[(size_t)(K + 1) * FF + col1];
        unsigned h0 = bf16_rne(f0), h1 = bf16_rne(f1);
        bh[2 * j] = (unsigned short)(h0 >> 16);
        bh[2 * j + 1] = (unsigned short)(h1 >> 16);
        bl[2 * j] = (unsigned short)(bf16_rne(f0 - __uint_as_float(h0)) >> 16);
        bl[2 * j + 1] = (unsigned short)(bf16_rne(f1 - __uint_as_float(h1)) >> 16);
      }
      c1 = WMMA_BF16(ah, bh, c1);
      c1 = WMMA_BF16(ah, bl, c1);
      c1 = WMMA_BF16(al, bh, c1);
    }
    {
      float bias = b1[e * FF + col1];
#pragma unroll
      for (int r = 0; r < 8; ++r) {
        int M = r + 8 * g;
        sH[M * FC + wave * 16 + ln] = pack_split(fmaxf(c1[r] + bias, 0.f));
      }
    }
    __syncthreads();

#pragma unroll 1
    for (int kk = 0; kk < FC; kk += 32) {
      v16u ah, al;
#pragma unroll
      for (int i = 0; i < 8; ++i) {
        int K = kk + 16 * g + 2 * i;
        unsigned w0 = sH[ln * FC + K];
        unsigned w1 = sH[ln * FC + K + 1];
        ah[2 * i] = (unsigned short)(w0 >> 16); al[2 * i] = (unsigned short)(w0 & 0xffffu);
        ah[2 * i + 1] = (unsigned short)(w1 >> 16); al[2 * i + 1] = (unsigned short)(w1 & 0xffffu);
      }
#pragma unroll
      for (int tt = 0; tt < 8; ++tt) {
        v16u bh, bl;
        int col2 = wave * 128 + tt * 16 + ln;
#pragma unroll
        for (int j = 0; j < 8; ++j) {
          int K = ff0 + kk + 16 * g + 2 * j;
          float f0 = W2p[(size_t)K * H + col2];
          float f1 = W2p[(size_t)(K + 1) * H + col2];
          unsigned h0 = bf16_rne(f0), h1 = bf16_rne(f1);
          bh[2 * j] = (unsigned short)(h0 >> 16);
          bh[2 * j + 1] = (unsigned short)(h1 >> 16);
          bl[2 * j] = (unsigned short)(bf16_rne(f0 - __uint_as_float(h0)) >> 16);
          bl[2 * j + 1] = (unsigned short)(bf16_rne(f1 - __uint_as_float(h1)) >> 16);
        }
        acc[tt] = WMMA_BF16(ah, bh, acc[tt]);
        acc[tt] = WMMA_BF16(ah, bl, acc[tt]);
        acc[tt] = WMMA_BF16(al, bh, acc[tt]);
      }
    }
    __syncthreads();
  }

#pragma unroll
  for (int tt = 0; tt < 8; ++tt) {
    int col = wave * 128 + tt * 16 + ln;
    float bias = b2[e * H + col];
#pragma unroll
    for (int r = 0; r < 8; ++r) {
      int M = r + 8 * g;
      int tkn = sTok[M];
      if (tkn >= 0) {
        float v = (acc[tt][r] + bias) * sW[M];
        atomicAdd(&out[(size_t)tkn * H + col], v);
      }
    }
  }
}

// -------------------------------------------------------------- launch ----
extern "C" void kernel_launch(void* const* d_in, const int* in_sizes, int n_in,
                              void* d_out, int out_size, void* d_ws, size_t ws_size,
                              hipStream_t stream) {
  (void)in_sizes; (void)n_in; (void)out_size;
  const float* x  = (const float*)d_in[0];
  const float* Wg = (const float*)d_in[1];
  const float* bg = (const float*)d_in[2];
  const float* W1 = (const float*)d_in[3];
  const float* b1 = (const float*)d_in[4];
  const float* W2 = (const float*)d_in[5];
  const float* b2 = (const float*)d_in[6];

  float* out    = (float*)d_out;
  float* logits = out + (size_t)T * H;

  char* ws = (char*)d_ws;
  int*   counts  = (int*)ws;
  int*   tok_idx = (int*)(ws + 256);
  float* tok_w   = (float*)(ws + 256 + (size_t)NE * T * 4);

  const size_t listBytes  = 256 + 2 * (size_t)NE * T * 4;         // 524544
  const size_t planeElems = (size_t)NE * H * FF;                  // 33.5M
  const size_t planeBytes = planeElems * 2;                       // 67 MB
  const size_t needBytes  = listBytes + 4 * planeBytes;           // ~269 MB

  const int nOut = T * H;
  moe_init<<<(nOut + 255) / 256, 256, 0, stream>>>(out, nOut, counts);
  moe_gate<<<T / 8, 256, 0, stream>>>(x, Wg, bg, logits, counts, tok_idx, tok_w);

  if (ws_size >= needBytes) {
    unsigned short* W1hi = (unsigned short*)(ws + listBytes);
    unsigned short* W1lo = (unsigned short*)(ws + listBytes + planeBytes);
    unsigned short* W2hi = (unsigned short*)(ws + listBytes + 2 * planeBytes);
    unsigned short* W2lo = (unsigned short*)(ws + listBytes + 3 * planeBytes);
    moe_pack<<<16384, 256, 0, stream>>>(W1, W2, W1hi, W1lo, W2hi, W2lo);
    moe_ffn_packed<<<NE * MAXT32, 256, 0, stream>>>(x, W1hi, W1lo, W2hi, W2lo,
                                                    b1, b2, counts, tok_idx, tok_w, out);
  } else {
    moe_ffn_inline<<<NE * MAXT16, 256, 0, stream>>>(x, W1, b1, W2, b2,
                                                    counts, tok_idx, tok_w, out);
  }
}
